// MarginLoss_24300924961023
// MI455X (gfx1250) — matrix operand order repaired and checked by reading, not compiled, not run on hardware
//
#include <hip/hip_runtime.h>

#define NN 8192
#define DD 128
#define KTR 7
#define NK (NN*KTR)
#define MARGINF 0.2f
#define THRF 0.5f
#define BIGF 1.0e6f
#define INFF 3.0e38f

#define CSPLIT 4                 // column quarters -> 2048 waves total
#define COLSQ (NN / CSPLIT)      // 2048 cols per wave
#define STRIPC 32                // columns per LDS-staged strip
#define NSTRIP (COLSQ / STRIPC)  // 64 strips
#define LISTS (CSPLIT * 2)       // sorted 7-lists per row
#define BPAD 136                 // padded halves per staged column (128+8)

typedef _Float16 h16;
typedef __attribute__((ext_vector_type(16))) _Float16 v16h;
typedef __attribute__((ext_vector_type(8)))  _Float16 v8h;
typedef __attribute__((ext_vector_type(8)))  float    v8f;
typedef __attribute__((ext_vector_type(4)))  int      v4i;

// ---------------------------------------------------------------------------
// CDNA5 async global->LDS copy (ASYNCcnt-tracked) + wait
// builtin signature (from clang diagnostic): (v4i AS1*, v4i AS3*, Ii, Ii)
// ---------------------------------------------------------------------------
__device__ __forceinline__ void async_b128(const h16* g, h16* l) {
#if __has_builtin(__builtin_amdgcn_global_load_async_to_lds_b128)
    __builtin_amdgcn_global_load_async_to_lds_b128(
        (__attribute__((address_space(1))) v4i*)(uintptr_t)g,
        (__attribute__((address_space(3))) v4i*)(unsigned)(uintptr_t)l,
        0, 0);
#else
    asm volatile("global_load_async_to_lds_b128 %0, %1, off"
                 :: "v"((unsigned)(uintptr_t)l),
                    "v"((unsigned long long)(uintptr_t)g)
                 : "memory");
#endif
}

#if __has_builtin(__builtin_amdgcn_s_wait_asynccnt)
#define ASYNC_WAIT(n) __builtin_amdgcn_s_wait_asynccnt(n)
#else
#define ASYNC_WAIT(n) asm volatile("s_wait_asynccnt %0" :: "n"(n) : "memory")
#endif

// ---------------------------------------------------------------------------
// Zero the scalar accumulators (harness poisons ws; re-zero every call)
// ---------------------------------------------------------------------------
__global__ void zero_kernel(float* accf, unsigned* accu) {
    if (threadIdx.x < 8) accf[threadIdx.x] = 0.0f;
    if (threadIdx.x == 8) accu[0] = 0u;
}

// ---------------------------------------------------------------------------
// Split f32 embeddings into hi/lo f16 halves; compute row squared norms.
// ---------------------------------------------------------------------------
__global__ __launch_bounds__(128) void prep_kernel(const float* __restrict__ x,
                                                   h16* __restrict__ xh,
                                                   h16* __restrict__ xl,
                                                   float* __restrict__ sq) {
    int row = blockIdx.x;
    int t   = threadIdx.x;
    float v = x[(size_t)row * DD + t];
    h16 hv = (h16)v;
    xh[(size_t)row * DD + t] = hv;
    xl[(size_t)row * DD + t] = (h16)(v - (float)hv);
    float p = v * v;
#pragma unroll
    for (int off = 16; off > 0; off >>= 1) p += __shfl_xor(p, off, 32);
    __shared__ float red[4];
    int w = threadIdx.x >> 5, l = threadIdx.x & 31;
    if (l == 0) red[w] = p;
    __syncthreads();
    if (threadIdx.x == 0) sq[row] = red[0] + red[1] + red[2] + red[3];
}

// ---------------------------------------------------------------------------
// Fused split-f16 WMMA GEMM + distance + negative mining.
// Block (8 waves) shares one column quarter; B strips are double-buffered in
// LDS via async global->LDS copies; each wave owns a 16-row strip.
// ---------------------------------------------------------------------------
__global__ __launch_bounds__(256) void gemm_mine_kernel(
    const h16* __restrict__ xh, const h16* __restrict__ xl,
    const float* __restrict__ sq,
    float* __restrict__ cand_s, int* __restrict__ cand_i,
    float* __restrict__ accf, unsigned* __restrict__ accu)
{
    __shared__ h16 Bh[2][STRIPC * BPAD];   // staged hi halves (padded cols)
    __shared__ h16 Bl[2][STRIPC * BPAD];   // staged lo halves
    __shared__ float Smine[8][16][33];     // per-wave 16x32 score tile (+pad)

    const int tid  = threadIdx.x;
    const int wave = tid >> 5;
    const int lane = tid & 31;
    const int quarter = blockIdx.x & (CSPLIT - 1);
    const int rowblk  = blockIdx.x / CSPLIT;
    const int r0  = (rowblk * 8 + wave) * 16;
    const int m   = lane & 15;
    const int hi  = lane >> 4;
    const int cq0 = quarter * COLSQ;

    // async-copy assignment: thread -> (column-in-strip, 16-half chunk)
    const int cc = tid >> 3;         // 0..31
    const int ko = (tid & 7) * 16;   // 0..112

    auto issue_strip = [&](int buf, int cb) {
        const h16* gh = xh + (size_t)(cb + cc) * DD + ko;
        const h16* gl = xl + (size_t)(cb + cc) * DD + ko;
        h16* lh = &Bh[buf][cc * BPAD + ko];
        h16* ll = &Bl[buf][cc * BPAD + ko];
        async_b128(gh, lh);
        async_b128(gh + 8, lh + 8);
        async_b128(gl, ll);
        async_b128(gl + 8, ll + 8);
    };

    // --- A fragments (rows r0..r0+15), hi/lo split, resident for whole loop.
    // 16-bit A 16x32 layout: lanes<16 hold K {0..7,16..23}; lanes>=16 hold
    // K {8..15,24..31} for row M=lane&15.
    v16h ah[4], al[4];
#pragma unroll
    for (int ks = 0; ks < 4; ++ks) {
        const h16* pa = xh + (size_t)(r0 + m) * DD + ks * 32 + hi * 8;
        const h16* pl = xl + (size_t)(r0 + m) * DD + ks * 32 + hi * 8;
        union { v16h v; v8h h[2]; } ua, ul;
        ua.h[0] = *(const v8h*)(pa);
        ua.h[1] = *(const v8h*)(pa + 16);
        ul.h[0] = *(const v8h*)(pl);
        ul.h[1] = *(const v8h*)(pl + 16);
        ah[ks] = ua.v;
        al[ks] = ul.v;
    }
    float rs[8];
#pragma unroll
    for (int v = 0; v < 8; ++v) rs[v] = sq[r0 + hi * 8 + v];

    float bs[KTR]; int bi[KTR];
#pragma unroll
    for (int q = 0; q < KTR; ++q) { bs[q] = INFF; bi[q] = 0; }
    float    dsum = 0.0f;
    unsigned ncnt = 0u;

    issue_strip(0, cq0);                       // prologue: stage strip 0

    for (int s = 0; s < NSTRIP; ++s) {
        const int nb    = s & 1;
        const int cbase = cq0 + s * STRIPC;
        if (s + 1 < NSTRIP) {
            issue_strip(nb ^ 1, cbase + STRIPC);   // prefetch next strip
            ASYNC_WAIT(4);                         // current strip's 4 done
        } else {
            ASYNC_WAIT(0);
        }
        __syncthreads();                           // strip visible block-wide

        // two 16x16 tiles from the staged strip
#pragma unroll
        for (int tt = 0; tt < 2; ++tt) {
            const int cl = tt * 16 + m;            // this lane's B column
            const h16* bph = &Bh[nb][cl * BPAD + hi * 8];
            const h16* bpl = &Bl[nb][cl * BPAD + hi * 8];
            v8f acc = {};
#pragma unroll
            for (int ks = 0; ks < 4; ++ks) {
                // B 32x16 layout: lanes<16 K {0..7,16..23}, lanes>=16 K {8..15,24..31}
                union { v16h v; v8h h[2]; } ubh, ubl;
                ubh.h[0] = *(const v8h*)(bph + ks * 32);
                ubh.h[1] = *(const v8h*)(bph + ks * 32 + 16);
                ubl.h[0] = *(const v8h*)(bpl + ks * 32);
                ubl.h[1] = *(const v8h*)(bpl + ks * 32 + 16);
                acc = __builtin_amdgcn_wmma_f32_16x16x32_f16(
                    false, ah[ks], false, ubh.v, (short)0, acc, false, false);
                acc = __builtin_amdgcn_wmma_f32_16x16x32_f16(
                    false, ah[ks], false, ubl.v, (short)0, acc, false, false);
                acc = __builtin_amdgcn_wmma_f32_16x16x32_f16(
                    false, al[ks], false, ubh.v, (short)0, acc, false, false);
            }
            const int   bcol = cbase + cl;
            const float cq   = sq[bcol];
            const int   ccls = bcol >> 3;
#pragma unroll
            for (int v = 0; v < 8; ++v) {
                const int grow = r0 + hi * 8 + v;  // C layout: M = v + 8*hi
                float d2   = rs[v] + cq - 2.0f * acc[v];
                d2         = fmaxf(d2, 0.0f);
                float dist = sqrtf(fmaxf(d2, 1e-4f));
                bool pos = ((grow >> 3) == ccls) && (grow != bcol);
                if (!pos) dsum += dist;            // nonpos includes diagonal
                bool nid = pos || (dist < THRF);
                ncnt += nid ? 1u : 0u;
                Smine[wave][v + hi * 8][tt * 16 + m] = nid ? (dist + BIGF) : dist;
            }
        }
        // mine this 16x32 strip (wave-private LDS; DS ops are in-order per
        // wave so the transposed reads see this wave's writes)
        {
#pragma unroll 4
            for (int j = 0; j < 16; ++j) {
                int   c  = hi * 16 + j;
                float sc = Smine[wave][m][c];
                if (sc < bs[KTR - 1]) {
                    bs[KTR - 1] = sc; bi[KTR - 1] = cbase + c;
#pragma unroll
                    for (int q = KTR - 1; q > 0; --q) {
                        if (bs[q] < bs[q - 1]) {
                            float tf = bs[q]; bs[q] = bs[q - 1]; bs[q - 1] = tf;
                            int   ti = bi[q]; bi[q] = bi[q - 1]; bi[q - 1] = ti;
                        }
                    }
                }
            }
        }
        __syncthreads();   // all waves done with buffer nb before reuse
    }

    // emit this lane's sorted 7-list: 8 lists per row (4 quarters x 2 lanes)
    {
        const int row   = r0 + m;
        const int listb = (row * LISTS + quarter * 2 + hi) * KTR;
#pragma unroll
        for (int q = 0; q < KTR; ++q) { cand_s[listb + q] = bs[q]; cand_i[listb + q] = bi[q]; }
    }
    // wave-reduce scalar partials, one atomic per wave
#pragma unroll
    for (int off = 16; off > 0; off >>= 1) {
        dsum += __shfl_xor(dsum, off, 32);
        ncnt += (unsigned)__shfl_xor((int)ncnt, off, 32);
    }
    if (lane == 0) {
        atomicAdd(&accf[0], dsum);
        atomicAdd(accu, ncnt);
    }
}

// ---------------------------------------------------------------------------
// Merge the 8 sorted 7-lists per row -> 7 smallest, then sort ascending by
// column index (reference's nonzero() order). Fully register-resident.
// ---------------------------------------------------------------------------
__global__ __launch_bounds__(256) void merge_kernel(
    const float* __restrict__ cand_s, const int* __restrict__ cand_i,
    int* __restrict__ neg_idx)
{
    const int CAND = LISTS * KTR;   // 56
    int row = blockIdx.x * 256 + threadIdx.x;
    if (row >= NN) return;
    float ss[CAND]; int ii[CAND];
#pragma unroll
    for (int q = 0; q < CAND; ++q) { ss[q] = cand_s[row * CAND + q]; ii[q] = cand_i[row * CAND + q]; }
    int picked[KTR];
#pragma unroll
    for (int k = 0; k < KTR; ++k) {
        float mv = INFF; int mi = 0; int mp = -1;
#pragma unroll
        for (int q = 0; q < CAND; ++q) {
            bool b = ss[q] < mv;
            mv = b ? ss[q] : mv;
            mi = b ? ii[q] : mi;
            mp = b ? q : mp;
        }
        picked[k] = mi;
#pragma unroll
        for (int q = 0; q < CAND; ++q) if (q == mp) ss[q] = INFF;
    }
#pragma unroll
    for (int a = 0; a < KTR; ++a)
#pragma unroll
        for (int b2 = 0; b2 < KTR - 1 - a; ++b2)
            if (picked[b2] > picked[b2 + 1]) {
                int t = picked[b2]; picked[b2] = picked[b2 + 1]; picked[b2 + 1] = t;
            }
#pragma unroll
    for (int k = 0; k < KTR; ++k) neg_idx[row * KTR + k] = picked[k];
}

// ---------------------------------------------------------------------------
// Triplet distances (full f32) + loss accumulation. One wave per anchor.
// pos_idx is structural: labels = arange//8 -> j-th positive of anchor a is
// base + j + (j >= a-base).
// ---------------------------------------------------------------------------
__global__ __launch_bounds__(256) void triplet_kernel(
    const float* __restrict__ x, const float* __restrict__ beta,
    const int* __restrict__ neg_idx, float* __restrict__ out,
    float* __restrict__ accf)
{
    const int wave = threadIdx.x >> 5, lane = threadIdx.x & 31;
    const int a = blockIdx.x * 8 + wave;
    const float4 av = *(const float4*)(x + (size_t)a * DD + lane * 4);
    const float bta = beta[0];
    const int base = (a >> 3) << 3;
    const int aoff = a - base;
    float lsum = 0.f, lnrof = 0.f, lcp = 0.f, lcn = 0.f;
    for (int k = 0; k < KTR; ++k) {
        int p  = base + k + (k >= aoff ? 1 : 0);
        int ng = neg_idx[a * KTR + k];
        float4 pv = *(const float4*)(x + (size_t)p * DD + lane * 4);
        float4 nv = *(const float4*)(x + (size_t)ng * DD + lane * 4);
        float sap, san, spn;
        {
            float d0 = pv.x - av.x, d1 = pv.y - av.y, d2 = pv.z - av.z, d3 = pv.w - av.w;
            sap = d0 * d0 + d1 * d1 + d2 * d2 + d3 * d3;
        }
        {
            float d0 = nv.x - av.x, d1 = nv.y - av.y, d2 = nv.z - av.z, d3 = nv.w - av.w;
            san = d0 * d0 + d1 * d1 + d2 * d2 + d3 * d3;
        }
        {
            float d0 = pv.x - nv.x, d1 = pv.y - nv.y, d2 = pv.z - nv.z, d3 = pv.w - nv.w;
            spn = d0 * d0 + d1 * d1 + d2 * d2 + d3 * d3;
        }
#pragma unroll
        for (int off = 16; off > 0; off >>= 1) {
            sap += __shfl_xor(sap, off, 32);
            san += __shfl_xor(san, off, 32);
            spn += __shfl_xor(spn, off, 32);
        }
        float d_ap = sqrtf(sap + 1e-8f);
        float d_an = sqrtf(san + 1e-8f);
        float d_pn = sqrtf(spn + 1e-8f);
        if (lane == 0) {
            int o = a * KTR + k;
            out[1 + o]          = d_ap;
            out[1 + NK + o]     = d_an;
            out[1 + 2 * NK + o] = d_pn;
            float pl = fmaxf(d_ap - bta + MARGINF, 0.f);
            float nl = fmaxf(bta - d_an + MARGINF, 0.f);
            lsum  += pl + nl;
            lnrof += (pl > 0.f || nl > 0.f) ? 1.f : 0.f;
            lcp   += (pl > 0.f) ? 1.f : 0.f;
            lcn   += (nl > 0.f) ? 1.f : 0.f;
        }
    }
    if (lane == 0) {
        atomicAdd(&accf[2], lsum);
        atomicAdd(&accf[3], lnrof);
        atomicAdd(&accf[4], lcp);
        atomicAdd(&accf[5], lcn);
    }
}

// ---------------------------------------------------------------------------
// Scalars. nonpos count = N*(N-7) = 8192*8185 = 67051520 (incl. diagonal).
// ---------------------------------------------------------------------------
__global__ void finalize_kernel(const float* __restrict__ beta,
                                const float* __restrict__ accf,
                                const unsigned* __restrict__ accu,
                                float* __restrict__ out)
{
    float sum_dist = accf[0];
    float sum_loss = accf[2], nrof = accf[3], cp = accf[4], cn = accf[5];
    float t = (float)NK;
    out[0]              = (nrof > 0.f) ? (sum_loss / nrof) : sum_loss;
    out[1 + 3 * NK + 0] = cp / t;                         // frac pos
    out[1 + 3 * NK + 1] = cn / t;                         // frac neg
    out[1 + 3 * NK + 2] = t;                              // t
    out[1 + 3 * NK + 3] = sum_dist / 67051520.0f;         // total_neg_dist
    out[1 + 3 * NK + 4] = nrof;                           // nrof
    out[1 + 3 * NK + 5] = beta[0];                        // beta
    out[1 + 3 * NK + 6] = ((float)accu[0]) / sum_dist;    // unused_neg_part
}

// ---------------------------------------------------------------------------
extern "C" void kernel_launch(void* const* d_in, const int* in_sizes, int n_in,
                              void* d_out, int out_size, void* d_ws, size_t ws_size,
                              hipStream_t stream)
{
    (void)in_sizes; (void)n_in; (void)out_size; (void)ws_size;
    const float* x    = (const float*)d_in[0];
    const float* beta = (const float*)d_in[1];
    float* out = (float*)d_out;

    const size_t CAND = (size_t)LISTS * KTR;              // 56 per row
    char* w = (char*)d_ws;
    h16*   xh     = (h16*)(w);                            // 2 MB
    h16*   xl     = (h16*)(w + 2u * NN * DD);             // 2 MB
    float* sq     = (float*)(w + 4u * NN * DD);           // 32 KB
    float* cand_s = (float*)(w + 4u * NN * DD + 4u * NN);
    int*   cand_i = (int*)  (w + 4u * NN * DD + 4u * NN + 4u * NN * CAND);
    int*   neg    = (int*)  (w + 4u * NN * DD + 4u * NN + 8u * NN * CAND);
    float* accf   = (float*)(w + 4u * NN * DD + 4u * NN + 8u * NN * CAND + 4u * NN * KTR);
    unsigned* accu = (unsigned*)(accf + 8);

    zero_kernel<<<1, 32, 0, stream>>>(accf, accu);
    prep_kernel<<<NN, 128, 0, stream>>>(x, xh, xl, sq);
    gemm_mine_kernel<<<(NN / 128) * CSPLIT, 256, 0, stream>>>(xh, xl, sq, cand_s, cand_i, accf, accu);
    merge_kernel<<<NN / 256, 256, 0, stream>>>(cand_s, cand_i, neg);
    triplet_kernel<<<NN / 8, 256, 0, stream>>>(x, beta, neg, out, accf);
    finalize_kernel<<<1, 1, 0, stream>>>(beta, accf, accu, out);
}
